// AttentionModel_39848706573130
// MI455X (gfx1250) — matrix operand order repaired and checked by reading, not compiled, not run on hardware
//
#include <hip/hip_runtime.h>

// ws requirement: 3 * B*H*S*E * 2 bytes = 48 MB (f16 Q, K, V^T intermediates)

typedef _Float16 v16h __attribute__((ext_vector_type(16)));
typedef _Float16 v8h  __attribute__((ext_vector_type(8)));
typedef float    v8f  __attribute__((ext_vector_type(8)));
typedef unsigned u32x4 __attribute__((ext_vector_type(4)));
typedef int      i32x4 __attribute__((ext_vector_type(4)));
typedef int      i32x8 __attribute__((ext_vector_type(8)));

#define NB 2
#define NS 2048
#define NH 16
#define NE 128
#define SCALE 0.08838834764831845f  // 1/sqrt(128)

union H16 { v16h v; v8h h[2]; };
static __device__ __forceinline__ v16h comb(v8h a, v8h b) { H16 u; u.h[0] = a; u.h[1] = b; return u.v; }

// TDM: issue one tensor_load_to_lds from descriptor groups 0/1 (2D tile).
// Groups 2/3 zero => tile_dim2/3/4 = 0 (unused). cpol = 0.
static __device__ __forceinline__ void tdm_load(u32x4 g0, i32x8 g1) {
  const i32x4 z4 = {0, 0, 0, 0};
#if __clang_major__ >= 23
  const i32x8 z8 = {0, 0, 0, 0, 0, 0, 0, 0};
  __builtin_amdgcn_tensor_load_to_lds(g0, g1, z4, z4, z8, 0);
#else
  __builtin_amdgcn_tensor_load_to_lds(g0, g1, z4, z4, 0);
#endif
}

// ---------------------------------------------------------------------------
// Kernel 1: fused QKV projection. X[b,s,h,:] @ W^T + b, fp32 -> fp16.
// Q,K out as [B,H,S,E] f16; V out transposed as [B,H,E,S] f16.
// ---------------------------------------------------------------------------
#define WP 136  // LDS pitch (halfs) for W tile: breaks 256B bank cycle

__global__ __launch_bounds__(256) void proj_kernel(
    const float* __restrict__ Xq, const float* __restrict__ Xk, const float* __restrict__ Xv,
    const float* __restrict__ Wq, const float* __restrict__ bq,
    const float* __restrict__ Wk, const float* __restrict__ bk,
    const float* __restrict__ Wv, const float* __restrict__ bv,
    _Float16* __restrict__ Qh, _Float16* __restrict__ Kh, _Float16* __restrict__ Vth)
{
  __shared__ _Float16 Wl[128 * WP];
  const int tid = threadIdx.x;

  const float* X; const float* W; const float* bias; _Float16* Out; int tr = 0;
  if (blockIdx.y == 0)      { X = Xq; W = Wq; bias = bq; Out = Qh; }
  else if (blockIdx.y == 1) { X = Xk; W = Wk; bias = bk; Out = Kh; }
  else                      { X = Xv; W = Wv; bias = bv; Out = Vth; tr = 1; }

  // stage W (128x128 f32 -> f16 in LDS)
  for (int i = tid; i < 128 * 128; i += 256)
    Wl[(i >> 7) * WP + (i & 127)] = (_Float16)W[i];
  __syncthreads();

  const int lane = tid & 31, w = tid >> 5;
  const int m  = lane & 15;        // A-matrix row / B-matrix column within tile
  const int hi = lane >> 4;        // half-wave select
  const int kA = hi * 8;           // A fragment K base (16-bit A 16x32 layout)
  const int kb = hi * 8;           // B fragment K base (same K order as A)
  const int n  = m;
  const int bh = blockIdx.x >> 4;  // b*16+h
  const int b  = bh >> 4, h = bh & 15;
  const int s0 = (blockIdx.x & 15) * 128 + w * 16;

  // A fragments: 16 token rows x 128 (four K=32 chunks), fp32 -> fp16 on the fly
  v16h A[4];
  const size_t rowbase = (((size_t)(b * NS + s0 + m)) * NH + h) * NE;
#pragma unroll
  for (int c = 0; c < 4; ++c) {
    const float* px = X + rowbase + c * 32 + kA;
    v8f f0 = *(const v8f*)px;          // K = c*32+kA .. +7
    v8f f1 = *(const v8f*)(px + 16);   // K = c*32+kA+16 .. +23
    A[c] = comb(__builtin_convertvector(f0, v8h), __builtin_convertvector(f1, v8h));
  }

  const v8f vzero = {0.f, 0.f, 0.f, 0.f, 0.f, 0.f, 0.f, 0.f};

#pragma unroll
  for (int t = 0; t < 8; ++t) {       // 8 N-tiles of 16 features
    v16h Bf[4];
#pragma unroll
    for (int c = 0; c < 4; ++c) {
      const _Float16* pw = &Wl[(t * 16 + n) * WP + c * 32 + kb];
      Bf[c] = comb(*(const v8h*)pw, *(const v8h*)(pw + 16));
    }
    v8f acc = vzero;
#pragma unroll
    for (int c = 0; c < 4; ++c)
      acc = __builtin_amdgcn_wmma_f32_16x16x32_f16(false, A[c], false, Bf[c],
                                                   (short)0, acc, false, false);
    const float bia = bias[t * 16 + n];
    if (!tr) {
#pragma unroll
      for (int r = 0; r < 8; ++r)
        Out[((size_t)(bh * NS + s0 + hi * 8 + r)) * NE + t * 16 + n] =
            (_Float16)(acc[r] + bia);
    } else {
      v8h pk;
#pragma unroll
      for (int r = 0; r < 8; ++r) pk[r] = (_Float16)(acc[r] + bia);
      *(v8h*)&Out[((size_t)(bh * NE + t * 16 + n)) * NS + s0 + hi * 8] = pk;
    }
  }
}

// ---------------------------------------------------------------------------
// Kernel 2: flash attention. Per block: 128 query rows of one (b,h); stream
// keys/values in 32-wide chunks via double-buffered TDM tensor_load_to_lds
// (TENSORcnt), online softmax, fp16 P, WMMA f16 compute.
// ---------------------------------------------------------------------------
#define KP 136  // K chunk LDS pitch (halfs): TDM pad 64DW rows + 4DW
#define VP 40   // V^T chunk LDS pitch (halfs): TDM pad 16DW rows + 4DW
#define PP 40   // P scratch pitch (halfs)

__global__ __launch_bounds__(256) void fa_kernel(
    const _Float16* __restrict__ Qh, const _Float16* __restrict__ Kh,
    const _Float16* __restrict__ Vth, float* __restrict__ Out)
{
  __shared__ _Float16 Kl[2][32 * KP];     // double-buffered K[32][128]
  __shared__ _Float16 Vl[2][128 * VP];    // double-buffered V^T[128][32]
  __shared__ _Float16 Pl[8 * 16 * PP];    // per-wave P transpose scratch

  const int tid  = threadIdx.x;
  const int lane = tid & 31, w = tid >> 5;
  const int m  = lane & 15;
  const int hi = lane >> 4;
  const int kA = hi * 8;
  const int kb = hi * 8;
  const int n  = m;
  const int bh = blockIdx.x;
  const int q0 = blockIdx.y * 128 + w * 16;

  // preload Q A-fragments for this wave's 16 rows (live across the loop)
  v16h Aq[4];
#pragma unroll
  for (int c = 0; c < 4; ++c) {
    const _Float16* pq = Qh + ((size_t)(bh * NS + q0 + m)) * NE + c * 32 + kA;
    Aq[c] = comb(*(const v8h*)pq, *(const v8h*)(pq + 16));
  }

  const v8f vzero = {0.f, 0.f, 0.f, 0.f, 0.f, 0.f, 0.f, 0.f};
  v8f o[8];
#pragma unroll
  for (int t = 0; t < 8; ++t) o[t] = vzero;
  float mrow[8], lrow[8];
#pragma unroll
  for (int r = 0; r < 8; ++r) { mrow[r] = -1e30f; lrow[r] = 0.f; }

  _Float16* Pw = &Pl[w * 16 * PP];

  // --- TDM descriptors (all fields block-uniform -> SGPRs) ---------------
  const _Float16* Kbase = Kh  + (size_t)bh * NS * NE;  // K rows [S][E]
  const _Float16* Vbase = Vth + (size_t)bh * NE * NS;  // V^T rows [E][S]
  const unsigned klds[2] = {(unsigned)(uintptr_t)&Kl[0][0], (unsigned)(uintptr_t)&Kl[1][0]};
  const unsigned vlds[2] = {(unsigned)(uintptr_t)&Vl[0][0], (unsigned)(uintptr_t)&Vl[1][0]};

  // group1 words (static): data_size=2B (code 1), pad_enable,
  //  K: pad_interval=5 (64 DW = one 256B row), pad_amount=3 (4 DW = 16B) -> pitch 272B
  //  V: pad_interval=3 (16 DW = one 64B row),  pad_amount=3 (4 DW = 16B) -> pitch 80B
  // [1]=tensor_dim0<<16, [2]=dim0_hi | dim1_lo<<16, [3]=dim1_hi | tile_dim0<<16,
  // [4]=tile_dim1, [5]=dim0_stride_lo, [6..7]=0
  const i32x8 g1k = {(int)((1u<<16)|(1u<<20)|(5u<<22)|(3u<<25)),
                     (int)(128u<<16), (int)(2048u<<16), (int)(128u<<16),
                     32, 128, 0, 0};
  const i32x8 g1v = {(int)((1u<<16)|(1u<<20)|(3u<<22)|(3u<<25)),
                     (int)(2048u<<16), (int)(128u<<16), (int)(32u<<16),
                     128, 2048, 0, 0};

#define TDM_ISSUE(buf, kcn)                                                     \
  {                                                                             \
    unsigned long long ka = (unsigned long long)(uintptr_t)(Kbase + (size_t)(kcn) * NE); \
    u32x4 g0k = {1u, klds[buf], (unsigned)ka, (unsigned)(ka >> 32) | 0x80000000u};       \
    tdm_load(g0k, g1k);                                                         \
    unsigned long long va = (unsigned long long)(uintptr_t)(Vbase + (kcn));     \
    u32x4 g0v = {1u, vlds[buf], (unsigned)va, (unsigned)(va >> 32) | 0x80000000u};       \
    tdm_load(g0v, g1v);                                                         \
  }

  if (tid < 32) TDM_ISSUE(0, 0);   // prologue: fill buffer 0
  int cur = 0;

  for (int kc = 0; kc < NS; kc += 32) {
    __builtin_amdgcn_s_wait_tensorcnt(0);  // buf[cur] DMA complete (issuing waves)
    __syncthreads();                       // publish buf[cur]; prev readers of buf[cur^1] done
    if (tid < 32 && kc + 32 < NS) TDM_ISSUE(cur ^ 1, kc + 32);  // overlap next DMA
    if (tid == 0 && kc + 64 < NS) {
      __builtin_prefetch(Kbase + (size_t)(kc + 64) * NE, 0, 0);
      __builtin_prefetch(Vbase + kc + 64, 0, 0);
    }

    const _Float16* Kc = Kl[cur];
    const _Float16* Vc = Vl[cur];

    // S = Q K^T for this 16x32 slab (two 16x16 tiles, K-dim = E in 4 chunks)
    v8f s0v = vzero, s1v = vzero;
#pragma unroll
    for (int c = 0; c < 4; ++c) {
      const _Float16* p0 = &Kc[n * KP + c * 32 + kb];         // cols kc..kc+15
      const _Float16* p1 = &Kc[(16 + n) * KP + c * 32 + kb];  // cols kc+16..kc+31
      v16h B0 = comb(*(const v8h*)p0, *(const v8h*)(p0 + 16));
      v16h B1 = comb(*(const v8h*)p1, *(const v8h*)(p1 + 16));
      s0v = __builtin_amdgcn_wmma_f32_16x16x32_f16(false, Aq[c], false, B0, (short)0, s0v, false, false);
      s1v = __builtin_amdgcn_wmma_f32_16x16x32_f16(false, Aq[c], false, B1, (short)0, s1v, false, false);
    }

    // online softmax: row stats live per-lane (vgpr r -> row r+8*hi)
    float al[8];
#pragma unroll
    for (int r = 0; r < 8; ++r) {
      float x0 = s0v[r] * SCALE, x1 = s1v[r] * SCALE;
      float tmax = fmaxf(x0, x1);
      tmax = fmaxf(tmax, __shfl_xor(tmax, 1, 32));
      tmax = fmaxf(tmax, __shfl_xor(tmax, 2, 32));
      tmax = fmaxf(tmax, __shfl_xor(tmax, 4, 32));
      tmax = fmaxf(tmax, __shfl_xor(tmax, 8, 32));  // stays within 16-lane half
      float mn = fmaxf(mrow[r], tmax);
      float a  = __expf(mrow[r] - mn);
      float p0 = __expf(x0 - mn);
      float p1 = __expf(x1 - mn);
      float rs = p0 + p1;
      rs += __shfl_xor(rs, 1, 32);
      rs += __shfl_xor(rs, 2, 32);
      rs += __shfl_xor(rs, 4, 32);
      rs += __shfl_xor(rs, 8, 32);
      lrow[r] = lrow[r] * a + rs;
      mrow[r] = mn;
      al[r]   = a;
      // P (f16) -> LDS in row-major [m][k] for the A-fragment of P@V
      Pw[(hi * 8 + r) * PP + n]      = (_Float16)p0;
      Pw[(hi * 8 + r) * PP + 16 + n] = (_Float16)p1;
    }
#pragma unroll
    for (int t = 0; t < 8; ++t)
#pragma unroll
      for (int r = 0; r < 8; ++r) o[t][r] *= al[r];

    __builtin_amdgcn_wave_barrier();  // DS ops are in-order per wave; fence scheduling

    // P@V: A = P (16x32 f16), B = V chunk (32x16 per N-tile) from V^T rows
    const _Float16* pp = &Pw[m * PP + kA];
    v16h Ap = comb(*(const v8h*)pp, *(const v8h*)(pp + 16));
#pragma unroll
    for (int t = 0; t < 8; ++t) {
      const _Float16* pv = &Vc[(t * 16 + n) * VP + kb];
      v16h Bv = comb(*(const v8h*)pv, *(const v8h*)(pv + 16));
      o[t] = __builtin_amdgcn_wmma_f32_16x16x32_f16(false, Ap, false, Bv, (short)0, o[t], false, false);
    }

    cur ^= 1;
  }

  // epilogue: O /= l, store fp32 [B,H,S,E]
#pragma unroll
  for (int r = 0; r < 8; ++r) {
    const float inv = 1.0f / lrow[r];
    const size_t ob = ((size_t)(bh * NS + q0 + hi * 8 + r)) * NE + n;
#pragma unroll
    for (int t = 0; t < 8; ++t)
      Out[ob + t * 16] = o[t][r] * inv;
  }
}

// ---------------------------------------------------------------------------
extern "C" void kernel_launch(void* const* d_in, const int* in_sizes, int n_in,
                              void* d_out, int out_size, void* d_ws, size_t ws_size,
                              hipStream_t stream) {
  const float* q  = (const float*)d_in[0];
  const float* k  = (const float*)d_in[1];
  const float* v  = (const float*)d_in[2];
  const float* Wq = (const float*)d_in[3];
  const float* bq = (const float*)d_in[4];
  const float* Wk = (const float*)d_in[5];
  const float* bk = (const float*)d_in[6];
  const float* Wv = (const float*)d_in[7];
  const float* bv = (const float*)d_in[8];

  const size_t tok = (size_t)NB * NH * NS * NE;
  _Float16* Qh  = (_Float16*)d_ws;
  _Float16* Kh  = Qh + tok;
  _Float16* Vth = Kh + tok;

  dim3 g1(NB * NH * (NS / 128), 3);
  proj_kernel<<<g1, 256, 0, stream>>>(q, k, v, Wq, bq, Wk, bk, Wv, bv, Qh, Kh, Vth);

  dim3 g2(NB * NH, NS / 128);
  fa_kernel<<<g2, 256, 0, stream>>>(Qh, Kh, Vth, (float*)d_out);
}
